// GeniePath_60163901882852
// MI455X (gfx1250) — matrix operand order repaired and checked by reading, not compiled, not run on hardware
//
#include <hip/hip_runtime.h>
#include <hip/hip_bf16.h>

// ---------------------------------------------------------------------------
// GeniePath on gfx1250 (CDNA5). All dense GEMMs run on v_wmma_f32_16x16x32_f16
// with an f16 hi/lo split (3 WMMAs per 16x16x32 step) to keep ~fp32 accuracy.
// ---------------------------------------------------------------------------

typedef __attribute__((ext_vector_type(16))) _Float16 v16h;
typedef __attribute__((ext_vector_type(8)))  float    v8f;

#define HDIM  128
#define INDIM 256
#define ODIM  64
#define DEPTH 3

static __device__ __forceinline__ unsigned f2ord(float f) {
  unsigned u = __float_as_uint(f);
  return (u & 0x80000000u) ? ~u : (u | 0x80000000u);
}
static __device__ __forceinline__ float ord2f(unsigned k) {
  unsigned u = (k & 0x80000000u) ? (k & 0x7FFFFFFFu) : ~k;
  return __uint_as_float(u);
}
static __device__ __forceinline__ float sigmoidf_(float x) {
  return 1.0f / (1.0f + expf(-x));
}

// Convert W [IN][OC] f32 -> transposed f16 hi/lo [OC][IN]
__global__ void k_prep_w(const float* __restrict__ W, _Float16* __restrict__ hi,
                         _Float16* __restrict__ lo, int IN, int OC) {
  int idx = blockIdx.x * blockDim.x + threadIdx.x;
  if (idx >= IN * OC) return;
  int i = idx / OC, o = idx % OC;
  float w = W[idx];
  _Float16 h = (_Float16)w;
  hi[(size_t)o * IN + i] = h;
  lo[(size_t)o * IN + i] = (_Float16)(w - (float)h);
}

__global__ void k_zero(float* __restrict__ p, long n) {
  long i = (long)blockIdx.x * blockDim.x + threadIdx.x;
  if (i < n) p[i] = 0.0f;
}

// C[M,OC] = act(A[M,K]f32 @ W(f16 hi/lo,[OC][K]) + bias).  One wave per 16x16
// tile, 8 waves per block.  ACT: 0=none, 1=relu.  Optional duplicate store C2.
template <int ACT>
__global__ __launch_bounds__(256) void
k_gemm(const float* __restrict__ A, const _Float16* __restrict__ Whi,
       const _Float16* __restrict__ Wlo, const float* __restrict__ bias,
       float* __restrict__ C, float* __restrict__ C2, int M, int K, int OC) {
  int ntn   = OC >> 4;
  int mt_n  = (M + 15) >> 4;
  int tile  = blockIdx.x * 8 + (threadIdx.x >> 5);
  if (tile >= mt_n * ntn) return;
  int mt = tile / ntn, nt = tile % ntn;
  int lane = threadIdx.x & 31;
  int r = lane & 15, hf = lane >> 4;
  int m0 = mt * 16, n0 = nt * 16;
  int mrow = m0 + r; if (mrow > M - 1) mrow = M - 1;
  const float*    arow = A   + (size_t)mrow * K;
  const _Float16* bh   = Whi + (size_t)(n0 + r) * K;
  const _Float16* bl   = Wlo + (size_t)(n0 + r) * K;
  v8f acc = {};
  for (int kk = 0; kk < K; kk += 32) {
    v16h ah, al;
#pragma unroll
    for (int half = 0; half < 2; ++half) {
      int kb = kk + 16 * hf + 8 * half;
#pragma unroll
      for (int t = 0; t < 8; ++t) {
        float av = arow[kb + t];
        _Float16 hv = (_Float16)av;
        ah[8 * half + t] = hv;
        al[8 * half + t] = (_Float16)(av - (float)hv);
      }
    }
    v16h bhv = *(const v16h*)(bh + kk + 16 * hf);
    v16h blv = *(const v16h*)(bl + kk + 16 * hf);
    acc = __builtin_amdgcn_wmma_f32_16x16x32_f16(false, ah, false, bhv, (short)0, acc, false, false);
    acc = __builtin_amdgcn_wmma_f32_16x16x32_f16(false, ah, false, blv, (short)0, acc, false, false);
    acc = __builtin_amdgcn_wmma_f32_16x16x32_f16(false, al, false, bhv, (short)0, acc, false, false);
  }
  float bv = bias ? bias[n0 + r] : 0.0f;
#pragma unroll
  for (int v = 0; v < 8; ++v) {
    int m = m0 + v + 8 * hf;
    if (m < M) {
      float val = acc[v] + bv;
      if (ACT == 1) val = fmaxf(val, 0.0f);
      size_t o = (size_t)m * OC + (n0 + r);
      C[o] = val;
      if (C2) C2[o] = val;
    }
  }
}

// el[n]=z[n,:].al ; er[n]=z[n,:].ar  (one wave per node)
__global__ void k_attn(const float* __restrict__ z, const float* __restrict__ al,
                       const float* __restrict__ ar, float* __restrict__ el,
                       float* __restrict__ er, int M) {
  int gid = blockIdx.x * blockDim.x + threadIdx.x;
  int n = gid >> 5, lane = gid & 31;
  if (n >= M) return;
  const float* zr = z + (size_t)n * HDIM;
  float sl = 0.f, sr = 0.f;
  for (int k = lane; k < HDIM; k += 32) {
    float v = zr[k];
    sl += v * al[k];
    sr += v * ar[k];
  }
  for (int o = 16; o; o >>= 1) {
    sl += __shfl_xor(sl, o, 32);
    sr += __shfl_xor(sr, o, 32);
  }
  if (lane == 0) { el[n] = sl; er[n] = sr; }
}

__global__ void k_init_node(unsigned* __restrict__ emaxu, float* __restrict__ denom, int M) {
  int i = blockIdx.x * blockDim.x + threadIdx.x;
  if (i < M) { emaxu[i] = 0x007FFFFFu; /* f2ord(-inf) */ denom[i] = 0.0f; }
}

__global__ void k_edge1(const int* __restrict__ src, const int* __restrict__ dst,
                        const float* __restrict__ el, const float* __restrict__ er,
                        float* __restrict__ ev, unsigned* __restrict__ emaxu, int E) {
  int e = blockIdx.x * blockDim.x + threadIdx.x;
  if (e >= E) return;
  float v = el[src[e]] + er[dst[e]];
  v = v > 0.0f ? v : 0.2f * v;          // leaky_relu(0.2)
  ev[e] = v;
  atomicMax(&emaxu[dst[e]], f2ord(v));
}

__global__ void k_emax_fix(const unsigned* __restrict__ emaxu, float* __restrict__ emaxf, int M) {
  int i = blockIdx.x * blockDim.x + threadIdx.x;
  if (i >= M) return;
  float m = ord2f(emaxu[i]);
  emaxf[i] = isfinite(m) ? m : 0.0f;    // guard isolated nodes
}

__global__ void k_edge2(const int* __restrict__ dst, float* __restrict__ ev,
                        const float* __restrict__ emaxf, float* __restrict__ denom, int E) {
  int e = blockIdx.x * blockDim.x + threadIdx.x;
  if (e >= E) return;
  float ex = expf(ev[e] - emaxf[dst[e]]);
  ev[e] = ex;
  atomicAdd(&denom[dst[e]], ex);
}

// agg[dst] += alpha * z[src]; 128 lanes per edge, 2 edges per 256-thread block
__global__ void k_edge3(const int* __restrict__ src, const int* __restrict__ dst,
                        const float* __restrict__ ev, const float* __restrict__ denom,
                        const float* __restrict__ z, float* __restrict__ agg, int E) {
  int e = blockIdx.x * 2 + (threadIdx.x >> 7);
  int hh = threadIdx.x & 127;
  if (e >= E) return;
  int d = dst[e], s = src[e];
  float alpha = ev[e] / fmaxf(denom[d], 1e-16f);
  atomicAdd(&agg[(size_t)d * HDIM + hh], alpha * z[(size_t)s * HDIM + hh]);
}

__global__ void k_bfin(const float* __restrict__ agg, const float* __restrict__ b,
                       float* __restrict__ h, long total) {
  long i = (long)blockIdx.x * blockDim.x + threadIdx.x;
  if (i < total) h[i] = tanhf(agg[i] + b[i & (HDIM - 1)]);
}

// Fused LSTM depth step: for a 16-row tile, compute ig/fg/og/ct via WMMA over
// hm=[coll|mu] (K=256), then update c and mu in place.
// Whi/Wlo layout: [4 gates][HDIM][2*HDIM] (gate order ig,fg,og,st).
__global__ __launch_bounds__(256) void
k_lstm(const float* __restrict__ coll, float* __restrict__ mu, float* __restrict__ c,
       const _Float16* __restrict__ Whi, const _Float16* __restrict__ Wlo,
       const float* __restrict__ igb, const float* __restrict__ fgb,
       const float* __restrict__ ogb, const float* __restrict__ stb, int M) {
  __shared__ _Float16 aHi[16][2 * HDIM];
  __shared__ _Float16 aLo[16][2 * HDIM];
  __shared__ float gbuf[4][16][HDIM];
  int m0 = blockIdx.x * 16;
  int tid = threadIdx.x;
  // stage A = [coll | mu] rows m0..m0+15, f32 -> f16 hi/lo
  for (int idx = tid; idx < 16 * 2 * HDIM; idx += 256) {
    int rr = idx >> 8, kc = idx & 255;
    int row = m0 + rr; if (row > M - 1) row = M - 1;
    float v = (kc < HDIM) ? coll[(size_t)row * HDIM + kc]
                          : mu[(size_t)row * HDIM + (kc - HDIM)];
    _Float16 hv = (_Float16)v;
    aHi[rr][kc] = hv;
    aLo[rr][kc] = (_Float16)(v - (float)hv);
  }
  __syncthreads();
  int w = tid >> 5, lane = tid & 31, r = lane & 15, hf = lane >> 4;
#pragma unroll
  for (int it = 0; it < 4; ++it) {
    int l = w * 4 + it;          // 32 tiles: 4 gates x 8 column tiles
    int g = l >> 3, nt = l & 7;
    int n0 = nt * 16;
    const _Float16* bh = Whi + ((size_t)g * HDIM + n0 + r) * (2 * HDIM);
    const _Float16* bl = Wlo + ((size_t)g * HDIM + n0 + r) * (2 * HDIM);
    v8f acc = {};
    for (int kk = 0; kk < 2 * HDIM; kk += 32) {
      v16h ah, al;
#pragma unroll
      for (int half = 0; half < 2; ++half) {
        int kb = kk + 16 * hf + 8 * half;
#pragma unroll
        for (int t = 0; t < 8; ++t) {
          ah[8 * half + t] = aHi[r][kb + t];
          al[8 * half + t] = aLo[r][kb + t];
        }
      }
      v16h bhv = *(const v16h*)(bh + kk + 16 * hf);
      v16h blv = *(const v16h*)(bl + kk + 16 * hf);
      acc = __builtin_amdgcn_wmma_f32_16x16x32_f16(false, ah, false, bhv, (short)0, acc, false, false);
      acc = __builtin_amdgcn_wmma_f32_16x16x32_f16(false, ah, false, blv, (short)0, acc, false, false);
      acc = __builtin_amdgcn_wmma_f32_16x16x32_f16(false, al, false, bhv, (short)0, acc, false, false);
    }
#pragma unroll
    for (int v = 0; v < 8; ++v) gbuf[g][v + 8 * hf][n0 + r] = acc[v];
  }
  __syncthreads();
  for (int idx = tid; idx < 16 * HDIM; idx += 256) {
    int rr = idx >> 7, col = idx & (HDIM - 1);
    int m = m0 + rr;
    if (m >= M) continue;
    size_t gi = (size_t)m * HDIM + col;
    float ig = sigmoidf_(gbuf[0][rr][col] + igb[col]);
    float fg = sigmoidf_(gbuf[1][rr][col] + fgb[col]);
    float og = sigmoidf_(gbuf[2][rr][col] + ogb[col]);
    float ct = tanhf(gbuf[3][rr][col] + stb[col]);
    float cn = fg * c[gi] + ig * ct;
    c[gi] = cn;
    mu[gi] = og * tanhf(cn);
  }
}

extern "C" void kernel_launch(void* const* d_in, const int* in_sizes, int n_in,
                              void* d_out, int out_size, void* d_ws, size_t ws_size,
                              hipStream_t stream) {
  const float* x    = (const float*)d_in[0];
  const int*   src  = (const int*)d_in[1];
  const int*   dst  = (const int*)d_in[2];
  const float* wxW  = (const float*)d_in[3];
  const float* wxb  = (const float*)d_in[4];
  const float* gatW = (const float*)d_in[5];
  const float* gatb = (const float*)d_in[6];
  const float* attl = (const float*)d_in[7];
  const float* attr = (const float*)d_in[8];
  const float* igW  = (const float*)d_in[9];
  const float* igb  = (const float*)d_in[10];
  const float* fgW  = (const float*)d_in[11];
  const float* fgb  = (const float*)d_in[12];
  const float* ogW  = (const float*)d_in[13];
  const float* ogb  = (const float*)d_in[14];
  const float* stW  = (const float*)d_in[15];
  const float* stb  = (const float*)d_in[16];
  const float* outW = (const float*)d_in[17];
  const float* outb = (const float*)d_in[18];
  float* out = (float*)d_out;

  const int M = in_sizes[0] / INDIM;   // 50000 nodes
  const int E = in_sizes[1];           // 800000 edges
  const int Mt = (M + 15) / 16;

  // f16 weight pool element offsets
  const size_t OFF_WX    = 0;                                   // [128][256]
  const size_t OFF_GAT   = OFF_WX + (size_t)HDIM * INDIM;       // 3 x [128][128]
  const size_t OFF_GATES = OFF_GAT + (size_t)DEPTH * HDIM * HDIM; // 3 x 4 x [128][256]
  const size_t OFF_OUT   = OFF_GATES + (size_t)DEPTH * 4 * HDIM * 2 * HDIM; // [64][128]
  const size_t WTOT      = OFF_OUT + (size_t)ODIM * HDIM;

  char* ws = (char*)d_ws;
  size_t off = 0;
  auto alloc = [&](size_t bytes) {
    off = (off + 255) & ~(size_t)255;
    size_t o = off; off += bytes; return o;
  };
  _Float16* whi  = (_Float16*)(ws + alloc(WTOT * 2));
  _Float16* wlo  = (_Float16*)(ws + alloc(WTOT * 2));
  float* hbuf    = (float*)(ws + alloc((size_t)M * HDIM * 4));
  float* mubuf   = (float*)(ws + alloc((size_t)M * HDIM * 4));
  float* cbuf    = (float*)(ws + alloc((size_t)M * HDIM * 4));
  float* zbuf    = (float*)(ws + alloc((size_t)M * HDIM * 4));
  float* aggbuf  = (float*)(ws + alloc((size_t)M * HDIM * 4));
  float* el      = (float*)(ws + alloc((size_t)M * 4));
  float* er      = (float*)(ws + alloc((size_t)M * 4));
  float* emaxf   = (float*)(ws + alloc((size_t)M * 4));
  float* denom   = (float*)(ws + alloc((size_t)M * 4));
  unsigned* emaxu = (unsigned*)(ws + alloc((size_t)M * 4));
  float* ev      = (float*)(ws + alloc((size_t)E * 4));

  auto cdiv = [](long a, long b) { return (int)((a + b - 1) / b); };

  // ---- weight prep: transpose + f16 hi/lo split ----
  k_prep_w<<<cdiv((long)INDIM * HDIM, 256), 256, 0, stream>>>(
      wxW, whi + OFF_WX, wlo + OFF_WX, INDIM, HDIM);
  for (int d = 0; d < DEPTH; ++d) {
    k_prep_w<<<cdiv((long)HDIM * HDIM, 256), 256, 0, stream>>>(
        gatW + (size_t)d * HDIM * HDIM,
        whi + OFF_GAT + (size_t)d * HDIM * HDIM,
        wlo + OFF_GAT + (size_t)d * HDIM * HDIM, HDIM, HDIM);
    const float* gsrc[4] = {igW, fgW, ogW, stW};
    for (int g = 0; g < 4; ++g) {
      size_t woff = OFF_GATES + ((size_t)d * 4 + g) * (size_t)HDIM * 2 * HDIM;
      k_prep_w<<<cdiv((long)2 * HDIM * HDIM, 256), 256, 0, stream>>>(
          gsrc[g] + (size_t)d * 2 * HDIM * HDIM, whi + woff, wlo + woff,
          2 * HDIM, HDIM);
    }
  }
  k_prep_w<<<cdiv((long)HDIM * ODIM, 256), 256, 0, stream>>>(
      outW, whi + OFF_OUT, wlo + OFF_OUT, HDIM, ODIM);

  // ---- h0 = x @ wx_W + wx_b  -> h and mu ----
  k_gemm<0><<<cdiv((long)Mt * (HDIM / 16), 8), 256, 0, stream>>>(
      x, whi + OFF_WX, wlo + OFF_WX, wxb, hbuf, mubuf, M, INDIM, HDIM);
  k_zero<<<cdiv((long)M * HDIM, 256), 256, 0, stream>>>(cbuf, (long)M * HDIM);

  // ---- depth loop: breadth (GAT) then fused LSTM step ----
  for (int d = 0; d < DEPTH; ++d) {
    k_gemm<0><<<cdiv((long)Mt * (HDIM / 16), 8), 256, 0, stream>>>(
        hbuf, whi + OFF_GAT + (size_t)d * HDIM * HDIM,
        wlo + OFF_GAT + (size_t)d * HDIM * HDIM, gatb + (size_t)d * HDIM,
        zbuf, nullptr, M, HDIM, HDIM);
    k_attn<<<cdiv((long)M * 32, 256), 256, 0, stream>>>(
        zbuf, attl + (size_t)d * HDIM, attr + (size_t)d * HDIM, el, er, M);
    k_init_node<<<cdiv(M, 256), 256, 0, stream>>>(emaxu, denom, M);
    k_zero<<<cdiv((long)M * HDIM, 256), 256, 0, stream>>>(aggbuf, (long)M * HDIM);
    k_edge1<<<cdiv(E, 256), 256, 0, stream>>>(src, dst, el, er, ev, emaxu, E);
    k_emax_fix<<<cdiv(M, 256), 256, 0, stream>>>(emaxu, emaxf, M);
    k_edge2<<<cdiv(E, 256), 256, 0, stream>>>(dst, ev, emaxf, denom, E);
    k_edge3<<<cdiv(E, 2), 256, 0, stream>>>(src, dst, ev, denom, zbuf, aggbuf, E);
    k_bfin<<<cdiv((long)M * HDIM, 256), 256, 0, stream>>>(
        aggbuf, gatb + (size_t)d * HDIM, hbuf, (long)M * HDIM);
    size_t gw = OFF_GATES + (size_t)d * 4 * HDIM * 2 * HDIM;
    k_lstm<<<Mt, 256, 0, stream>>>(hbuf, mubuf, cbuf, whi + gw, wlo + gw,
                                   igb + (size_t)d * HDIM, fgb + (size_t)d * HDIM,
                                   ogb + (size_t)d * HDIM, stb + (size_t)d * HDIM, M);
  }

  // ---- out = relu(mu @ out_W + out_b) ----
  k_gemm<1><<<cdiv((long)Mt * (ODIM / 16), 8), 256, 0, stream>>>(
      mubuf, whi + OFF_OUT, wlo + OFF_OUT, outb, out, nullptr, M, HDIM, ODIM);
}
